// QuantumLayer_49864570307124
// MI455X (gfx1250) — matrix operand order repaired and checked by reading, not compiled, not run on hardware
//
#include <hip/hip_runtime.h>

#define B_SZ   8192
#define IN_SZ  1024
#define OUT_SZ 1024
#define QBITS  10

typedef __attribute__((ext_vector_type(16))) __bf16 v16bf;
typedef __attribute__((ext_vector_type(8)))  float  v8f;
typedef __attribute__((ext_vector_type(4)))  unsigned int u32x4;
typedef __attribute__((ext_vector_type(8)))  int i32x8;
typedef __attribute__((ext_vector_type(4)))  int i32x4;

union Frag {
  uint4 q[2];
  v16bf v;
};

__device__ __forceinline__ unsigned short f2bf(float f) {
  unsigned int u = __float_as_uint(f);
  u += 0x7FFFu + ((u >> 16) & 1u);   // round-to-nearest-even
  return (unsigned short)(u >> 16);
}

__device__ __forceinline__ unsigned int pack2bf(float a, float b) {
  return (unsigned int)f2bf(a) | ((unsigned int)f2bf(b) << 16);
}

// ---------------------------------------------------------------------------
// TDM 2D tile load: global (bf16, row stride stride0 elements) -> LDS.
// Builds the D# per CDNA5 ISA 8.3/8.4: group0 = {count, lds_addr, global_addr,
// type=2}; group1 = {data_size=2B, tensor_dim0/1, tile_dim0/1, dim0 stride}.
// Groups 2/3 zero (2D tensor). Issued once per wave (EXEC ignored by TDM).
// ---------------------------------------------------------------------------
__device__ __forceinline__ void tdm_load_2d(unsigned int lds_off, const void* gptr,
                                            unsigned int tensor_d0, unsigned int tensor_d1,
                                            unsigned int tile_d0, unsigned int tile_d1,
                                            unsigned long long stride0) {
  const unsigned long long ga = (unsigned long long)(uintptr_t)gptr;
  u32x4 g0;
  g0.x = 0x1u;                                              // count = 1 (valid)
  g0.y = lds_off;                                           // LDS byte address
  g0.z = (unsigned int)(ga & 0xFFFFFFFFull);                // global_addr[31:0]
  g0.w = (unsigned int)((ga >> 32) & 0x1FFFFFFull) | (2u << 30);  // [56:32] | type=2
  i32x8 g1;
  g1[0] = (int)(1u << 16);                                  // data_size = 1 (2 bytes)
  g1[1] = (int)((tensor_d0 & 0xFFFFu) << 16);               // tensor_dim0[15:0]
  g1[2] = (int)((tensor_d0 >> 16) | ((tensor_d1 & 0xFFFFu) << 16));
  g1[3] = (int)((tensor_d1 >> 16) | (tile_d0 << 16));       // tile_dim0
  g1[4] = (int)tile_d1;                                     // tile_dim1, tile_dim2=0
  g1[5] = (int)(unsigned int)(stride0 & 0xFFFFFFFFull);     // tensor_dim0_stride
  g1[6] = (int)(unsigned int)((stride0 >> 32) & 0xFFFFull); // stride0 hi, stride1=0
  g1[7] = 0;
  i32x4 gz = {0, 0, 0, 0};
#if defined(__clang_major__) && (__clang_major__ >= 23)
  i32x8 gz8 = {0, 0, 0, 0, 0, 0, 0, 0};
  __builtin_amdgcn_tensor_load_to_lds(g0, g1, gz, gz, gz8, 0);
#else
  __builtin_amdgcn_tensor_load_to_lds(g0, g1, gz, gz, 0);
#endif
}

// ---------------------------------------------------------------------------
// Convert f32 -> bf16 (one float4 -> uint2 per thread).
// ---------------------------------------------------------------------------
__global__ __launch_bounds__(256)
void convert_kernel(const float* __restrict__ src, unsigned short* __restrict__ dst) {
  const size_t i4 = (size_t)blockIdx.x * blockDim.x + threadIdx.x;
  const float4 v = *(const float4*)(src + i4 * 4);
  *(uint2*)(dst + i4 * 4) = make_uint2(pack2bf(v.x, v.y), pack2bf(v.z, v.w));
}

// ---------------------------------------------------------------------------
// Kernel 0: tail vector (batch-constant columns 10..1023 of q) + its sq-norm.
// ---------------------------------------------------------------------------
__global__ __launch_bounds__(256)
void tail_kernel(const float* __restrict__ rotation,
                 float* __restrict__ tail, float* __restrict__ misc) {
  __shared__ float red[256];
  __shared__ float theta_s[QBITS];
  const int tid = threadIdx.x;
  if (tid < QBITS) theta_s[tid] = fmodf(rotation[tid], 6.2831853071795864f);
  __syncthreads();
  float ss = 0.0f;
  for (int t = tid; t < 1024; t += 256) {
    float v = 0.0f;
    if (t < IN_SZ - QBITS) {
      int i = t + QBITS;
      v = sinf(theta_s[i % QBITS] * (float)(i / QBITS + 1));
    }
    tail[t] = v;
    ss += v * v;
  }
  red[tid] = ss;
  __syncthreads();
  for (int s2 = 128; s2 > 0; s2 >>= 1) {
    if (tid < s2) red[tid] += red[tid + s2];
    __syncthreads();
  }
  if (tid == 0) misc[0] = red[0];
}

// ---------------------------------------------------------------------------
// Kernel 0b: T[n] = sum_j tail[j]*post_W[n,10+j]; bf16 B operand for GEMM2.
// ---------------------------------------------------------------------------
__global__ __launch_bounds__(256)
void bq_kernel(const float* __restrict__ postW, const float* __restrict__ tail,
               unsigned short* __restrict__ BqT) {
  const int n = blockIdx.x * blockDim.x + threadIdx.x;
  if (n >= OUT_SZ) return;
  const float* wrow = postW + (size_t)n * IN_SZ;
  float T = 0.0f;
  for (int j = 0; j < IN_SZ - QBITS; ++j) T += tail[j] * wrow[QBITS + j];
  unsigned short* o = BqT + (size_t)n * 32;
  for (int k = 0; k < 32; ++k) {
    float v = (k < QBITS) ? wrow[k] : (k == QBITS ? T : 0.0f);
    o[k] = f2bf(v);
  }
}

// ---------------------------------------------------------------------------
// Kernel 1: h = relu(x @ pre_W^T + pre_b). bf16 WMMA, f32 accumulate.
// Block tile 128x64, K-step 32. Tiles staged HBM/L2 -> LDS by the Tensor Data
// Mover (double buffered, wave 0 issues, TENSORcnt sync), no VGPR bounce.
// ---------------------------------------------------------------------------
__global__ __launch_bounds__(256)
void gemm1_kernel(const unsigned short* __restrict__ xb,
                  const unsigned short* __restrict__ Wb,
                  const float* __restrict__ bias, float* __restrict__ h) {
  __shared__ __align__(16) unsigned short As[2][128][32];   // 16 KB
  __shared__ __align__(16) unsigned short Bs[2][64][32];    //  8 KB

  const int tid  = threadIdx.x;
  const int lane = tid & 31;
  const int wave = tid >> 5;
  const int wm   = wave >> 1;            // 0..3
  const int wn   = wave & 1;             // 0..1
  const int blockM = blockIdx.y * 128;
  const int blockN = blockIdx.x * 64;
  const int l15 = lane & 15;
  const int ks  = (lane < 16) ? 0 : 8;   // A-fragment K-chunk base (ISA 7.12.2)
  const int kb  = (lane < 16) ? 0 : 16;  // unused; B-fragment loads use ks like A

  const unsigned int asOff = (unsigned int)(uintptr_t)&As[0][0][0];
  const unsigned int bsOff = (unsigned int)(uintptr_t)&Bs[0][0][0];
  const unsigned short* aG = xb + (size_t)blockM * IN_SZ;
  const unsigned short* bG = Wb + (size_t)blockN * IN_SZ;

  v8f acc[2][2] = {};

  if (wave == 0) {  // prologue: DMA tile 0 into buffer 0
    tdm_load_2d(asOff, aG, IN_SZ, B_SZ, 32, 128, IN_SZ);
    tdm_load_2d(bsOff, bG, IN_SZ, OUT_SZ, 32, 64, IN_SZ);
  }

  const int NSTEP = IN_SZ / 32;
  for (int it = 0; it < NSTEP; ++it) {
    const int cur = it & 1;
    if (wave == 0) {
      if (it + 1 < NSTEP) {
        const int nxt = (it + 1) & 1;
        tdm_load_2d(asOff + nxt * 8192u, aG + (it + 1) * 32, IN_SZ, B_SZ, 32, 128, IN_SZ);
        tdm_load_2d(bsOff + nxt * 4096u, bG + (it + 1) * 32, IN_SZ, OUT_SZ, 32, 64, IN_SZ);
        __builtin_amdgcn_s_wait_tensorcnt(2);   // tile `it` landed (in-order)
      } else {
        __builtin_amdgcn_s_wait_tensorcnt(0);
      }
    }
    __syncthreads();                            // publish buffer `cur`

    Frag a[2], b[2];
    #pragma unroll
    for (int mi = 0; mi < 2; ++mi) {
      const int r = wm * 32 + mi * 16 + l15;
      a[mi].q[0] = *(const uint4*)&As[cur][r][ks];
      a[mi].q[1] = *(const uint4*)&As[cur][r][ks + 16];
    }
    #pragma unroll
    for (int ni = 0; ni < 2; ++ni) {
      const int n = wn * 32 + ni * 16 + l15;
      b[ni].q[0] = *(const uint4*)&Bs[cur][n][ks];
      b[ni].q[1] = *(const uint4*)&Bs[cur][n][ks + 16];
    }
    #pragma unroll
    for (int mi = 0; mi < 2; ++mi)
      #pragma unroll
      for (int ni = 0; ni < 2; ++ni)
        acc[mi][ni] = __builtin_amdgcn_wmma_f32_16x16x32_bf16(
            false, a[mi].v, false, b[ni].v, (short)0, acc[mi][ni], false, false);
    __syncthreads();                            // done reading before overwrite
  }

  // Epilogue: +bias, relu, f32 store. D layout: VGPR i -> M=i / M=8+i.
  #pragma unroll
  for (int mi = 0; mi < 2; ++mi) {
    const int rbase = blockM + wm * 32 + mi * 16 + ((lane < 16) ? 0 : 8);
    #pragma unroll
    for (int ni = 0; ni < 2; ++ni) {
      const int col = blockN + wn * 32 + ni * 16 + l15;
      const float bb = bias[col];
      #pragma unroll
      for (int i = 0; i < 8; ++i) {
        float v = acc[mi][ni][i] + bb;
        h[(size_t)(rbase + i) * IN_SZ + col] = fmaxf(v, 0.0f);
      }
    }
  }
}

// ---------------------------------------------------------------------------
// Kernel 2: per batch row -- normalize, 10-stage butterfly in LDS, decoherence-
// damped probabilities, signed reduce -> head, final q normalization. Phases
// (rotation/entangle) cancel in |amp|^2. Emits 11-feature bf16 A-row (K=32).
// ---------------------------------------------------------------------------
__global__ __launch_bounds__(256)
void quantum_kernel(const float* __restrict__ h, const float* __restrict__ coherence,
                    const float* __restrict__ misc, unsigned short* __restrict__ Aq) {
  __shared__ float s[1024];
  __shared__ float red[256];
  __shared__ float wsum[8 * QBITS];
  __shared__ float hfin[QBITS];
  __shared__ float sinv_s;

  const int tid = threadIdx.x;
  const int row = blockIdx.x;
  const float* hr = h + (size_t)row * IN_SZ;

  float vloc[4];
  float ss = 0.0f;
  #pragma unroll
  for (int i = 0; i < 4; ++i) {
    float v = hr[tid + 256 * i];
    vloc[i] = v;
    ss += v * v;
  }
  red[tid] = ss;
  __syncthreads();
  for (int s2 = 128; s2 > 0; s2 >>= 1) {
    if (tid < s2) red[tid] += red[tid + s2];
    __syncthreads();
  }
  const float inv = (red[0] > 0.0f) ? rsqrtf(red[0]) : 1.0f;
  #pragma unroll
  for (int i = 0; i < 4; ++i) s[tid + 256 * i] = vloc[i] * inv;
  __syncthreads();

  const float RS2 = 0.70710678118654752f;
  #pragma unroll
  for (int j = 0; j < QBITS; ++j) {
    #pragma unroll
    for (int pp = 0; pp < 2; ++pp) {
      const int p = tid + 256 * pp;                 // pair index 0..511
      const int mask = (1 << j) - 1;
      const int i0 = ((p & ~mask) << 1) | (p & mask);
      const int i1 = i0 | (1 << j);
      const float a0 = s[i0], a1 = s[i1];
      s[i0] = (a0 - a1) * RS2;
      s[i1] = (a0 + a1) * RS2;
    }
    __syncthreads();
  }

  const float r = 1.0f / (1.0f + expf(-coherence[0]));
  float hl[QBITS];
  #pragma unroll
  for (int q = 0; q < QBITS; ++q) hl[q] = 0.0f;
  #pragma unroll
  for (int i = 0; i < 4; ++i) {
    const int d = tid + 256 * i;
    const float amp = s[d];
    const int pc = __popc(d);
    const float dec = (pc > 1) ? (1.0f - r * (float)pc * 0.1f) : 1.0f;
    const float p = amp * amp * dec * dec;
    #pragma unroll
    for (int q = 0; q < QBITS; ++q)
      hl[q] += ((d >> q) & 1) ? p : -p;
  }
  #pragma unroll
  for (int q = 0; q < QBITS; ++q) {
    float v = hl[q];
    #pragma unroll
    for (int off = 16; off > 0; off >>= 1) v += __shfl_xor(v, off, 32);
    if ((tid & 31) == 0) wsum[(tid >> 5) * QBITS + q] = v;
  }
  __syncthreads();
  if (tid < QBITS) {
    float v = 0.0f;
    for (int w = 0; w < 8; ++w) v += wsum[w * QBITS + tid];
    hfin[tid] = v * (1.0f - r);
  }
  __syncthreads();
  if (tid == 0) {
    float hs = 0.0f;
    for (int q = 0; q < QBITS; ++q) hs += hfin[q] * hfin[q];
    sinv_s = 1.0f / sqrtf(hs + misc[0]);           // tail norm > 0 always
  }
  __syncthreads();
  if (tid < 32) {
    const float sv = sinv_s;
    const float v = (tid < QBITS) ? hfin[tid] * sv : (tid == QBITS ? sv : 0.0f);
    Aq[(size_t)row * 32 + tid] = f2bf(v);
  }
}

// ---------------------------------------------------------------------------
// Kernel 3: collapsed GEMM2: out = Aq(11 live features, K=32) x BqT + post_b.
// ---------------------------------------------------------------------------
__global__ __launch_bounds__(256)
void gemm2_kernel(const unsigned short* __restrict__ Aq,
                  const unsigned short* __restrict__ BqT,
                  const float* __restrict__ post_b,
                  float* __restrict__ out) {
  const int tid  = threadIdx.x;
  const int lane = tid & 31;
  const int wave = tid >> 5;
  const int tile = blockIdx.x * 8 + wave;
  const int tm = tile >> 6;                  // OUT/16 = 64 column tiles
  const int tn = tile & 63;
  const int l15 = lane & 15;
  const int ks  = (lane < 16) ? 0 : 8;
  const int kb  = (lane < 16) ? 0 : 16;

  Frag a, b;
  const unsigned short* ar = Aq + (size_t)(tm * 16 + l15) * 32;
  a.q[0] = *(const uint4*)(ar + ks);
  a.q[1] = *(const uint4*)(ar + ks + 16);
  const unsigned short* br = BqT + (size_t)(tn * 16 + l15) * 32;
  b.q[0] = *(const uint4*)(br + ks);
  b.q[1] = *(const uint4*)(br + ks + 16);

  v8f acc = {};
  acc = __builtin_amdgcn_wmma_f32_16x16x32_bf16(
      false, a.v, false, b.v, (short)0, acc, false, false);

  const int col = tn * 16 + l15;
  const float bb = post_b[col];
  const int rbase = tm * 16 + ((lane < 16) ? 0 : 8);
  #pragma unroll
  for (int i = 0; i < 8; ++i)
    out[(size_t)(rbase + i) * OUT_SZ + col] = acc[i] + bb;
}

// ---------------------------------------------------------------------------
extern "C" void kernel_launch(void* const* d_in, const int* in_sizes, int n_in,
                              void* d_out, int out_size, void* d_ws, size_t ws_size,
                              hipStream_t stream) {
  (void)in_sizes; (void)n_in; (void)out_size; (void)ws_size;
  const float* x         = (const float*)d_in[0];
  const float* pre_W     = (const float*)d_in[1];
  const float* pre_b     = (const float*)d_in[2];
  const float* rotation  = (const float*)d_in[3];
  // d_in[4] = entangle: pure phase, cancels in |amplitude|^2 -> unused.
  const float* coherence = (const float*)d_in[5];
  const float* post_W    = (const float*)d_in[6];
  const float* post_b    = (const float*)d_in[7];
  float* out = (float*)d_out;

  char* ws = (char*)d_ws;
  float*          h    = (float*)(ws);                        // 32 MB
  unsigned short* xb   = (unsigned short*)(ws + 33554432);    // 16 MB
  unsigned short* Wb   = (unsigned short*)(ws + 50331648);    //  2 MB
  unsigned short* Aq   = (unsigned short*)(ws + 52428800);    // 512 KB
  unsigned short* BqT  = (unsigned short*)(ws + 52953088);    //  64 KB
  float*          tail = (float*)(ws + 53018624);             //   4 KB
  float*          misc = (float*)(ws + 53022720);

  convert_kernel<<<(B_SZ * IN_SZ / 4) / 256, 256, 0, stream>>>(x, xb);
  convert_kernel<<<(IN_SZ * IN_SZ / 4) / 256, 256, 0, stream>>>(pre_W, Wb);
  tail_kernel<<<1, 256, 0, stream>>>(rotation, tail, misc);
  bq_kernel<<<OUT_SZ / 256, 256, 0, stream>>>(post_W, tail, BqT);
  gemm1_kernel<<<dim3(IN_SZ / 64, B_SZ / 128), 256, 0, stream>>>(xb, Wb, pre_b, h);
  quantum_kernel<<<B_SZ, 256, 0, stream>>>(h, coherence, misc, Aq);
  gemm2_kernel<<<(B_SZ / 16) * (OUT_SZ / 16) / 8, 256, 0, stream>>>(Aq, BqT, post_b, out);
}
